// Attention_13219909337782
// MI455X (gfx1250) — matrix operand order repaired and checked by reading, not compiled, not run on hardware
//
#include <hip/hip_runtime.h>
#include <hip/hip_bf16.h>
#include <math.h>

// Attention forward: out = softmax(mask(QK^T/sqrt(D))) @ V, also returns p_attn.
// B=4, H=12, S=2048, D=64, fp32 in/out. d_out = [out (B*H*S*D) | p_attn (B*H*S*S)].

typedef __attribute__((ext_vector_type(16))) _Float16 v16h;
typedef __attribute__((ext_vector_type(4)))  _Float16 v4h;
typedef __attribute__((ext_vector_type(8)))  float    v8f;

#define BB 4
#define HN 12
#define SQ 2048
#define DD 64
// softmax done in exp2 domain: scores scaled by log2(e)/sqrt(D)
#define SCALE2 0.18033688011112042f   // 1.4426950408889634 / 8

// A-fragment K-index for VGPR-pair v (0..7), lane-half hi: {0..15} + 16*hi
__device__ __forceinline__ int akb(int v, int hi) {
    return 2 * v + hi * 16;
}

__device__ __forceinline__ v8f wmma_f16(v16h a, v16h b, v8f c) {
    return __builtin_amdgcn_wmma_f32_16x16x32_f16(false, a, false, b, (short)0, c,
                                                  false, false);
}

// cooperative f32 -> f16 tile stage: nf4 float4's, 256 threads, b128 loads + b64 LDS stores
__device__ __forceinline__ void stage_tile(_Float16* dst, const float* src, int nf4, int tid) {
    const float4* s4 = (const float4*)src;
    v4h* d4 = (v4h*)dst;
    #pragma unroll
    for (int i = tid; i < nf4; i += 256) {
        float4 f = s4[i];
        v4h h;
        h[0] = (_Float16)f.x; h[1] = (_Float16)f.y;
        h[2] = (_Float16)f.z; h[3] = (_Float16)f.w;
        d4[i] = h;
    }
}

__global__ __launch_bounds__(256) void attn_fwd_kernel(
    const float* __restrict__ Q, const float* __restrict__ K,
    const float* __restrict__ V, const int* __restrict__ mask,
    float* __restrict__ out, float* __restrict__ patt)
{
    __shared__ _Float16 kbuf[2][32][DD];     // double-buffered K tile (f16)
    __shared__ _Float16 vbuf[2][32][DD];     // double-buffered V tile (f16)
    __shared__ _Float16 pbuf[8][16][32];     // per-wave P tile C->A re-layout buffer

    const int tid  = threadIdx.x;
    const int wave = tid >> 5;
    const int lane = tid & 31;
    const int nl   = lane & 15;
    const int hi   = lane >> 4;

    const int bid  = blockIdx.x;
    const int qblk = bid & 15;               // S/128 = 16 q-blocks per head
    const int bh   = bid >> 4;               // b*H + h
    const int b    = bh / HN;
    const int q0   = qblk * 128 + wave * 16; // this wave's 16 query rows

    const float* Qh  = Q + (size_t)bh * SQ * DD;
    const float* Kh  = K + (size_t)bh * SQ * DD;
    const float* Vh  = V + (size_t)bh * SQ * DD;
    const int*   mb  = mask + b * SQ;
    float* outh = out  + (size_t)bh * SQ * DD;
    float* path = patt + (size_t)bh * SQ * SQ;

    // ---- Q A-fragments (f16), held in registers for both passes ----
    v16h aq0, aq1;
    {
        const float* qrow = Qh + (size_t)(q0 + nl) * DD;
        #pragma unroll
        for (int v = 0; v < 8; ++v) {
            int k = akb(v, hi);
            aq0[2 * v]     = (_Float16)qrow[k];
            aq0[2 * v + 1] = (_Float16)qrow[k + 1];
            aq1[2 * v]     = (_Float16)qrow[32 + k];
            aq1[2 * v + 1] = (_Float16)qrow[32 + k + 1];
        }
    }

    // per-row query mask (C-layout rows r + 8*hi)
    float mq[8];
    #pragma unroll
    for (int r = 0; r < 8; ++r) mq[r] = (float)mb[q0 + r + hi * 8];

    // per-LANE online softmax state, exp2 domain (no cross-lane ops in hot loop)
    float m[8], l[8];
    #pragma unroll
    for (int r = 0; r < 8; ++r) { m[r] = -INFINITY; l[r] = 0.0f; }

    // ================= pass 1: per-lane online softmax stats ================
    stage_tile(&kbuf[0][0][0], Kh, 32 * DD / 4, tid);
    for (int kp = 0; kp < SQ / 32; ++kp) {
        const int kb = kp * 32;
        const int db = kp & 1;
        __syncthreads();  // buffer db fully staged; buffer db^1 fully consumed
        if (kp + 1 < SQ / 32)
            stage_tile(&kbuf[db ^ 1][0][0], Kh + (size_t)(kb + 32) * DD, 32 * DD / 4, tid);
        if (kp + 2 < SQ / 32)
            __builtin_prefetch(Kh + (size_t)(kb + 64) * DD + tid * 8, 0, 1);

        #pragma unroll
        for (int hh = 0; hh < 2; ++hh) {
            const int kr = hh * 16 + nl;
            v16h b0, b1;
            #pragma unroll
            for (int v = 0; v < 8; ++v) {
                int d = hi * 16 + 2 * v;
                b0[2 * v]     = kbuf[db][kr][d];
                b0[2 * v + 1] = kbuf[db][kr][d + 1];
                b1[2 * v]     = kbuf[db][kr][32 + d];
                b1[2 * v + 1] = kbuf[db][kr][32 + d + 1];
            }
            v8f c = {};
            c = wmma_f16(aq0, b0, c);
            c = wmma_f16(aq1, b1, c);

            const float mk = (float)mb[kb + hh * 16 + nl];
            #pragma unroll
            for (int r = 0; r < 8; ++r) {
                float g  = mq[r] * mk;                 // 0 or 1
                float s  = __builtin_fmaf(c[r], g * SCALE2, (g - 1.0f) * 1e9f);
                float mn = fmaxf(m[r], s);             // per-lane online update
                l[r] = l[r] * __builtin_amdgcn_exp2f(m[r] - mn)
                            + __builtin_amdgcn_exp2f(s - mn);
                m[r] = mn;
            }
        }
    }

    // ---- one-time cross-lane (m,l) merge; fold log2(l) into the max ----
    #pragma unroll
    for (int r = 0; r < 8; ++r) {
        float mm = m[r], ll = l[r];
        #pragma unroll
        for (int off = 1; off < 16; off <<= 1) {
            float mo = __shfl_xor(mm, off);
            float lo = __shfl_xor(ll, off);
            float mn = fmaxf(mm, mo);
            ll = ll * __builtin_amdgcn_exp2f(mm - mn)
               + lo * __builtin_amdgcn_exp2f(mo - mn);
            mm = mn;
        }
        m[r] = mm + __builtin_amdgcn_logf(ll);  // p = exp2(s - m[r])
    }

    // ================= pass 2: emit p_attn once + PV accumulate =============
    v8f o[4] = {v8f{}, v8f{}, v8f{}, v8f{}};

    stage_tile(&kbuf[0][0][0], Kh, 32 * DD / 4, tid);
    stage_tile(&vbuf[0][0][0], Vh, 32 * DD / 4, tid);
    for (int kp = 0; kp < SQ / 32; ++kp) {
        const int kb = kp * 32;
        const int db = kp & 1;
        __syncthreads();
        if (kp + 1 < SQ / 32) {
            stage_tile(&kbuf[db ^ 1][0][0], Kh + (size_t)(kb + 32) * DD, 32 * DD / 4, tid);
            stage_tile(&vbuf[db ^ 1][0][0], Vh + (size_t)(kb + 32) * DD, 32 * DD / 4, tid);
        }

        // two 16-key halves: recompute scores, normalize, write p_attn, stage P
        #pragma unroll
        for (int hh = 0; hh < 2; ++hh) {
            const int kr = hh * 16 + nl;
            v16h b0, b1;
            #pragma unroll
            for (int v = 0; v < 8; ++v) {
                int d = hi * 16 + 2 * v;
                b0[2 * v]     = kbuf[db][kr][d];
                b0[2 * v + 1] = kbuf[db][kr][d + 1];
                b1[2 * v]     = kbuf[db][kr][32 + d];
                b1[2 * v + 1] = kbuf[db][kr][32 + d + 1];
            }
            v8f c = {};
            c = wmma_f16(aq0, b0, c);
            c = wmma_f16(aq1, b1, c);

            const float mk = (float)mb[kb + hh * 16 + nl];
            #pragma unroll
            for (int r = 0; r < 8; ++r) {
                float g = mq[r] * mk;
                float s = __builtin_fmaf(c[r], g * SCALE2, (g - 1.0f) * 1e9f);
                float p = __builtin_amdgcn_exp2f(s - m[r]);
                // streaming (non-temporal) write: p_attn is written once, never read
                __builtin_nontemporal_store(
                    p, &path[(size_t)(q0 + r + hi * 8) * SQ + kb + hh * 16 + nl]);
                pbuf[wave][r + hi * 8][hh * 16 + nl] = (_Float16)p;
            }
        }

        // P A-fragment (16 rows x 32 keys) from per-wave LDS staging
        v16h ap;
        #pragma unroll
        for (int v = 0; v < 8; ++v) {
            int k = akb(v, hi);
            ap[2 * v]     = pbuf[wave][nl][k];
            ap[2 * v + 1] = pbuf[wave][nl][k + 1];
        }
        // V B-fragments: N = d-slice, contraction = key
        #pragma unroll
        for (int n = 0; n < 4; ++n) {
            v16h bv;
            #pragma unroll
            for (int v = 0; v < 8; ++v) {
                int kr = hi * 16 + 2 * v;
                bv[2 * v]     = vbuf[db][kr][n * 16 + nl];
                bv[2 * v + 1] = vbuf[db][kr + 1][n * 16 + nl];
            }
            o[n] = wmma_f16(ap, bv, o[n]);
        }
    }

    // ---- write O (C layout: VGPR r -> row r + 8*hi, col = n*16 + nl) ----
    #pragma unroll
    for (int r = 0; r < 8; ++r) {
        float* orow = outh + (size_t)(q0 + r + hi * 8) * DD;
        __builtin_nontemporal_store(o[0][r], orow +  0 + nl);
        __builtin_nontemporal_store(o[1][r], orow + 16 + nl);
        __builtin_nontemporal_store(o[2][r], orow + 32 + nl);
        __builtin_nontemporal_store(o[3][r], orow + 48 + nl);
    }
}

extern "C" void kernel_launch(void* const* d_in, const int* in_sizes, int n_in,
                              void* d_out, int out_size, void* d_ws, size_t ws_size,
                              hipStream_t stream) {
    (void)in_sizes; (void)n_in; (void)out_size; (void)d_ws; (void)ws_size;
    const float* q    = (const float*)d_in[0];
    const float* k    = (const float*)d_in[1];
    const float* v    = (const float*)d_in[2];
    const int*   mask = (const int*)d_in[3];
    // d_in[4] = head_nums scalar (compile-time HN here)

    float* out  = (float*)d_out;
    float* patt = out + (size_t)BB * HN * SQ * DD;

    const int blocks = BB * HN * (SQ / 128);  // 768
    attn_fwd_kernel<<<blocks, 256, 0, stream>>>(q, k, v, mask, out, patt);
}
